// LAMForMTEBSuper_70205535420765
// MI455X (gfx1250) — compile-verified, matrix operand order checked by reading, not run on hardware
//
#include <hip/hip_runtime.h>
#include <hip/hip_bf16.h>
#include <math.h>

typedef __attribute__((ext_vector_type(16))) _Float16 v16h;
typedef __attribute__((ext_vector_type(8)))  float    v8f;

#define T_STEPS 16384
#define D_MODEL 384
#define NHEADS  12
#define DKV     32
#define FAST_DECAY 0.3f
#define SLOW_DECAY 0.85f

// ---------------------------------------------------------------------------
// Kernel 1: fused dual projection GEMM with WMMA f32_16x16x32_f16.
//   out[m, n] = sum_k hidden[m,k] * W[n,k]   (W = key_w or value_w, 384x384)
// Grid: (1024 M-tiles, 6).  blockIdx.y < 3 -> key path, else value path.
// 8 waves/block, each wave owns one 16x16 output tile; K fully unrolled
// (12 WMMAs) so the scheduler can pipeline global loads across WMMA issue.
// ---------------------------------------------------------------------------
__global__ __launch_bounds__(256, 2)
void lam_proj_wmma(const float* __restrict__ hidden,
                   const float* __restrict__ key_w,
                   const float* __restrict__ value_w,
                   float* __restrict__ kout,
                   float* __restrict__ vout) {
  const int wave = threadIdx.x >> 5;      // 0..7
  const int lane = threadIdx.x & 31;
  const int mTile = blockIdx.x;           // 0..1023
  int yy = blockIdx.y;                    // 0..5
  const float* __restrict__ W;
  float* __restrict__ out;
  if (yy < 3) { W = key_w;   out = kout; }
  else        { W = value_w; out = vout; yy -= 3; }
  const int nTile  = yy * 8 + wave;       // 0..23
  const int halfSel = lane >> 4;          // lane group: K-half select
  const int lid     = lane & 15;          // M (A) / N (B) index within tile

  const float* __restrict__ arow = hidden + (mTile * 16 + lid) * D_MODEL;
  const float* __restrict__ brow = W      + (nTile * 16 + lid) * D_MODEL;

  v8f acc = {};
#pragma unroll
  for (int k0 = 0; k0 < D_MODEL; k0 += 32) {
    const float* ap = arow + k0 + halfSel * 8;
    const float* bp = brow + k0 + halfSel * 8;
    v16h a, b;
#pragma unroll
    for (int i = 0; i < 8; ++i) {
      // VGPR pairs: [0..3] -> K = base+0..7, [4..7] -> K = base+16..23
      a[i]     = (_Float16)ap[i];
      a[i + 8] = (_Float16)ap[i + 16];
      b[i]     = (_Float16)bp[i];
      b[i + 8] = (_Float16)bp[i + 16];
    }
    acc = __builtin_amdgcn_wmma_f32_16x16x32_f16(
        /*neg_a=*/false, a, /*neg_b=*/false, b,
        /*c_mod=*/(short)0, acc, /*reuse_a=*/false, /*reuse_b=*/false);
  }
  // C/D layout: VGPR r, lanes 0-15: M=r, N=lane; lanes 16-31: M=r+8
#pragma unroll
  for (int r = 0; r < 8; ++r) {
    const int row = mTile * 16 + r + halfSel * 8;
    const int col = nTile * 16 + lid;
    out[row * D_MODEL + col] = acc[r];
  }
}

// ---------------------------------------------------------------------------
// Kernel 2: sequential dual-decay scan. 1 block, 12 wave32s, wave = head.
// State Wf/Ws [32x32] per head held in VGPRs: lane = column v, reg = row l.
// Per-step cross-head flux coupling via LDS + workgroup barrier.
// k-normalization (per-head L2 over DK=32 == wave width) fused at load.
// Final normalize(flatten(Ws)) fused at the end.
// ---------------------------------------------------------------------------
__global__ __launch_bounds__(NHEADS * 32, 1)
void lam_scan(const float* __restrict__ kproj,
              const float* __restrict__ vproj,
              float* __restrict__ out) {
  __shared__ float s_red[16];
  __shared__ float s_bcast;

  const int wave = threadIdx.x >> 5;   // head 0..11
  const int lane = threadIdx.x & 31;   // v index (and l index for k loads)

  float Wf[32], Ws[32];
#pragma unroll
  for (int i = 0; i < 32; ++i) { Wf[i] = 0.0f; Ws[i] = 0.0f; }

  const float* __restrict__ kbase = kproj + wave * DKV + lane;
  const float* __restrict__ vbase = vproj + wave * DKV + lane;

  for (int t = 0; t < T_STEPS; ++t) {
    const float kraw = kbase[t * D_MODEL];
    const float vt   = vbase[t * D_MODEL];

    // per-head L2 norm of k over the 32 lanes (butterfly reduce)
    float ss = kraw * kraw;
#pragma unroll
    for (int off = 16; off > 0; off >>= 1) ss += __shfl_xor(ss, off, 32);
    const float kn = kraw / fmaxf(sqrtf(ss), 1e-12f);

    // replicate kn[0..31] into every lane once (32 shuffles/step)
    float ka[32];
#pragma unroll
    for (int r = 0; r < 32; ++r) ka[r] = __shfl(kn, r, 32);

    // kW = k^T W  (per lane: its column's dot)
    float accF = 0.0f, accS = 0.0f;
#pragma unroll
    for (int l = 0; l < 32; ++l) {
      accF = fmaf(ka[l], Wf[l], accF);
      accS = fmaf(ka[l], Ws[l], accS);
    }
    // W' = decay*(W - k (k^T W)) + k v^T
#pragma unroll
    for (int l = 0; l < 32; ++l) {
      Wf[l] = FAST_DECAY * fmaf(-ka[l], accF, Wf[l]) + ka[l] * vt;
      Ws[l] = SLOW_DECAY * fmaf(-ka[l], accS, Ws[l]) + ka[l] * vt;
    }
    // v_fast/v_slow = W'^T k
    float vf = 0.0f, vs = 0.0f;
#pragma unroll
    for (int l = 0; l < 32; ++l) {
      vf = fmaf(ka[l], Wf[l], vf);
      vs = fmaf(ka[l], Ws[l], vs);
    }
    float d = vf - vs;
    float dsq = d * d;
#pragma unroll
    for (int off = 16; off > 0; off >>= 1) dsq += __shfl_xor(dsq, off, 32);
    if (lane == 0) s_red[wave] = dsq;
    __syncthreads();
    if (threadIdx.x == 0) {
      float tot = 0.0f;
#pragma unroll
      for (int h = 0; h < NHEADS; ++h) tot += s_red[h];
      const float dn = sqrtf(tot);
      s_bcast = 0.1f / (1.0f + __expf(-(dn - 0.5f)));  // sigmoid(dn-0.5)*0.1
    }
    __syncthreads();
    const float flux = s_bcast;
#pragma unroll
    for (int l = 0; l < 32; ++l) Ws[l] = fmaf(flux, Wf[l] - Ws[l], Ws[l]);
  }

  // normalize(flatten(Ws)) -> out[h*1024 + l*32 + v]
  float ss = 0.0f;
#pragma unroll
  for (int l = 0; l < 32; ++l) ss = fmaf(Ws[l], Ws[l], ss);
#pragma unroll
  for (int off = 16; off > 0; off >>= 1) ss += __shfl_xor(ss, off, 32);
  if (lane == 0) s_red[wave] = ss;
  __syncthreads();
  if (threadIdx.x == 0) {
    float tot = 0.0f;
#pragma unroll
    for (int h = 0; h < NHEADS; ++h) tot += s_red[h];
    s_bcast = 1.0f / fmaxf(sqrtf(tot), 1e-12f);
  }
  __syncthreads();
  const float inv = s_bcast;
#pragma unroll
  for (int l = 0; l < 32; ++l)
    out[wave * (DKV * DKV) + l * DKV + lane] = Ws[l] * inv;
}

// ---------------------------------------------------------------------------
extern "C" void kernel_launch(void* const* d_in, const int* in_sizes, int n_in,
                              void* d_out, int out_size, void* d_ws, size_t ws_size,
                              hipStream_t stream) {
  const float* hidden  = (const float*)d_in[0];   // [16384, 384]
  const float* key_w   = (const float*)d_in[1];   // [384, 384]
  const float* value_w = (const float*)d_in[2];   // [384, 384]
  float* out = (float*)d_out;                     // [12*32*32]

  float* kws = (float*)d_ws;                        // [16384, 384] raw k proj
  float* vws = kws + (size_t)T_STEPS * D_MODEL;     // [16384, 384] v proj

  dim3 g(T_STEPS / 16, 6);
  lam_proj_wmma<<<g, 256, 0, stream>>>(hidden, key_w, value_w, kws, vws);
  lam_scan<<<1, NHEADS * 32, 0, stream>>>(kws, vws, out);
}
